// EncoderBlock_86552180949462
// MI455X (gfx1250) — matrix operand order repaired and checked by reading, not compiled, not run on hardware
//
#include <hip/hip_runtime.h>
#include <math.h>

// Problem constants (match reference)
#define Bc   4
#define Sc   2048
#define Dc   1024
#define Hc   16
#define DFFc 4096
#define DKc  64
#define Mtok (Bc * Sc)   // 8192 token rows

typedef __bf16 bf16;
typedef __bf16 v16bf __attribute__((ext_vector_type(16)));
typedef __bf16 v8bf  __attribute__((ext_vector_type(8)));
typedef float  v8f   __attribute__((ext_vector_type(8)));
typedef int    v4i   __attribute__((ext_vector_type(4)));

typedef __attribute__((address_space(1))) v4i gbl_v4i;
typedef __attribute__((address_space(3))) v4i lds_v4i;

// Feature probes (device pass decides; host pass only parses)
#if defined(__has_builtin)
#if __has_builtin(__builtin_amdgcn_global_load_async_to_lds_b128)
#define HAVE_ASYNC_LDS 1
#endif
#endif
#ifndef HAVE_ASYNC_LDS
#define HAVE_ASYNC_LDS 0
#endif

template <int N>
__device__ __forceinline__ void wait_async() {
#if HAVE_ASYNC_LDS
#if __has_builtin(__builtin_amdgcn_s_wait_asynccnt)
  __builtin_amdgcn_s_wait_asynccnt(N);
#else
  asm volatile("s_wait_asynccnt %0" ::"i"(N) : "memory");
#endif
#endif
}

// ---------------------------------------------------------------------------
// WMMA helpers.  bf16 16x16x32, f32 accumulate.
// A-fragment (ISA 7.12.2, 16-bit A 16x32): lane (m = lane&15, h = lane>>4)
//   elems 0..7  = Arow[m][ 8*h ..  8*h+8)
//   elems 8..15 = Arow[m][16+8*h .. 16+8*h+8)
// B-fragment (K x 16, col n = lane&15): same K order as A-fragment.
// C layout: vgpr j holds row (j + 8*h), col = lane&15.
// ---------------------------------------------------------------------------
__device__ __forceinline__ v8f wmma_bf(v16bf a, v16bf b, v8f c) {
  return __builtin_amdgcn_wmma_f32_16x16x32_bf16(false, a, false, b, (short)0, c,
                                                 false, false);
}

__device__ __forceinline__ v16bf frag_a(const bf16* row, int hf) {
  v8bf lo = *(const v8bf*)(row + 8 * hf);
  v8bf hi = *(const v8bf*)(row + 16 + 8 * hf);
  v16bf r;
#pragma unroll
  for (int i = 0; i < 8; ++i) { r[i] = lo[i]; r[i + 8] = hi[i]; }
  return r;
}

__device__ __forceinline__ v16bf frag_b(const bf16* row, int hf) {
  v8bf lo = *(const v8bf*)(row + 8 * hf);
  v8bf hi = *(const v8bf*)(row + 16 + 8 * hf);
  v16bf r;
#pragma unroll
  for (int i = 0; i < 8; ++i) { r[i] = lo[i]; r[i + 8] = hi[i]; }
  return r;
}

__device__ __forceinline__ float redmax16(float v) {
#pragma unroll
  for (int m = 1; m < 16; m <<= 1) v = fmaxf(v, __shfl_xor(v, m, 32));
  return v;  // reduction stays within each 16-lane half
}
__device__ __forceinline__ float redsum16(float v) {
#pragma unroll
  for (int m = 1; m < 16; m <<= 1) v += __shfl_xor(v, m, 32);
  return v;
}

// ---------------------------------------------------------------------------
// fp32 -> bf16 convert
// ---------------------------------------------------------------------------
__global__ void cvt_bf16_kernel(const float* __restrict__ in, bf16* __restrict__ out,
                                int n) {
  int i = blockIdx.x * blockDim.x + threadIdx.x;
  if (i < n) out[i] = (bf16)in[i];
}

// ---------------------------------------------------------------------------
// LayerNorm (scalar gamma/beta, unbiased std, eps added to std) -> bf16
// one block (256 threads) per row of D=1024
// ---------------------------------------------------------------------------
__global__ void layernorm_kernel(const float* __restrict__ x,
                                 const float* __restrict__ gamma,
                                 const float* __restrict__ beta,
                                 bf16* __restrict__ out) {
  __shared__ float s1[256], s2[256];
  int row = blockIdx.x;
  const float* xr = x + (size_t)row * Dc;
  float a = 0.f, b = 0.f;
  for (int i = threadIdx.x; i < Dc; i += 256) { float v = xr[i]; a += v; b += v * v; }
  s1[threadIdx.x] = a; s2[threadIdx.x] = b;
  __syncthreads();
  for (int off = 128; off > 0; off >>= 1) {
    if (threadIdx.x < off) {
      s1[threadIdx.x] += s1[threadIdx.x + off];
      s2[threadIdx.x] += s2[threadIdx.x + off];
    }
    __syncthreads();
  }
  float mean = s1[0] / (float)Dc;
  float var  = (s2[0] - (float)Dc * mean * mean) / (float)(Dc - 1);  // ddof=1
  float inv  = 1.0f / (sqrtf(fmaxf(var, 0.f)) + 1e-6f);
  float g = gamma[0], bt = beta[0];
  bf16* orow = out + (size_t)row * Dc;
  for (int i = threadIdx.x; i < Dc; i += 256)
    orow[i] = (bf16)(g * (xr[i] - mean) * inv + bt);
}

// ---------------------------------------------------------------------------
// Generic WMMA GEMM:  C[M,N] = A[M,K] * W[N,K]^T  (+bias, +relu, +f32 residual)
// block = 256 threads (8 waves); block tile 256(M) x 64(N); wave tile 32x64.
// B tile (64 rows x 64 K) staged into double-buffered LDS via async
// global->LDS loads (ASYNCcnt); last iteration peeled so the hot loop is
// branch-free.  16 WMMA per wave between barrier pairs.
// mode 0: bf16 row-major   mode 1: f32 row-major
// mode 2: bf16 scatter -> [B,H,S,DK]     mode 3: bf16 scatter -> [B,H,DK,S]
// ---------------------------------------------------------------------------
__global__ void gemm_bf16_kernel(const bf16* __restrict__ A, const bf16* __restrict__ W,
                                 const float* __restrict__ bias,
                                 const float* __restrict__ resid,
                                 bf16* __restrict__ outb, float* __restrict__ outf,
                                 int M, int N, int K, int relu, int mode) {
  __shared__ bf16 Bs[2][64 * 64];  // double-buffered 64 W-rows x 64 K (16KB)
  const int wave = threadIdx.x >> 5;
  const int lane = threadIdx.x & 31;
  const int hf   = lane >> 4;
  const int n16  = lane & 15;
  const int n0    = blockIdx.x * 64;
  const int mBase = blockIdx.y * 256 + wave * 32;

  const bf16* arow0 = A + (size_t)(mBase + n16) * K;
  const bf16* arow1 = A + (size_t)(mBase + 16 + n16) * K;

  const v8f z = {0.f, 0.f, 0.f, 0.f, 0.f, 0.f, 0.f, 0.f};
  v8f acc[2][4] = {{z, z, z, z}, {z, z, z, z}};

  const int ldr = threadIdx.x >> 2;        // 0..63 : W row within tile
  const int ldc = (threadIdx.x & 3) * 8;   // 0,8,16,24 (+32 for 2nd chunk)
  const bf16* wsrc = W + (size_t)(n0 + ldr) * K + ldc;

  auto stage = [&](int buf, int kk) {
#if HAVE_ASYNC_LDS
    __builtin_amdgcn_global_load_async_to_lds_b128(
        (gbl_v4i*)(wsrc + kk),
        (lds_v4i*)(&Bs[buf][ldr * 64 + ldc]), 0, 0);
    __builtin_amdgcn_global_load_async_to_lds_b128(
        (gbl_v4i*)(wsrc + kk + 32),
        (lds_v4i*)(&Bs[buf][ldr * 64 + ldc + 32]), 0, 0);
#else
    *(v8bf*)(&Bs[buf][ldr * 64 + ldc])      = *(const v8bf*)(wsrc + kk);
    *(v8bf*)(&Bs[buf][ldr * 64 + ldc + 32]) = *(const v8bf*)(wsrc + kk + 32);
#endif
  };

  auto compute = [&](int cur, int kk) {
#pragma unroll
    for (int ks = 0; ks < 64; ks += 32) {
      v16bf a0 = frag_a(arow0 + kk + ks, hf);
      v16bf a1 = frag_a(arow1 + kk + ks, hf);
#pragma unroll
      for (int nt = 0; nt < 4; ++nt) {
        v16bf b = frag_b(&Bs[cur][(nt * 16 + n16) * 64 + ks], hf);
        acc[0][nt] = wmma_bf(a0, b, acc[0][nt]);
        acc[1][nt] = wmma_bf(a1, b, acc[1][nt]);
      }
    }
  };

  const int iters = K >> 6;
  stage(0, 0);
  for (int it = 0; it + 1 < iters; ++it) {
    const int kk = it << 6;
    stage((it + 1) & 1, kk + 64);     // prefetch next tile (async)
    wait_async<2>();                  // current tile fully landed (in-order)
    __syncthreads();
    __builtin_prefetch(arow0 + kk + 64, 0, 3);
    __builtin_prefetch(arow1 + kk + 64, 0, 3);
    compute(it & 1, kk);
    __syncthreads();                  // reads done before buffer reuse
  }
  wait_async<0>();
  __syncthreads();
  compute((iters - 1) & 1, (iters - 1) << 6);

  // epilogue
#pragma unroll
  for (int mi = 0; mi < 2; ++mi) {
#pragma unroll
    for (int nt = 0; nt < 4; ++nt) {
#pragma unroll
      for (int j = 0; j < 8; ++j) {
        int row = mBase + mi * 16 + j + hf * 8;
        int col = n0 + nt * 16 + n16;
        float v = acc[mi][nt][j];
        if (bias)  v += bias[col];
        if (relu)  v = fmaxf(v, 0.f);
        if (resid) v += resid[(size_t)row * N + col];
        if (mode == 0) {
          outb[(size_t)row * N + col] = (bf16)v;
        } else if (mode == 1) {
          outf[(size_t)row * N + col] = v;
        } else {
          int bb = row / Sc, ss = row % Sc;
          int hh = col >> 6, dk = col & 63;
          if (mode == 2)   // [B,H,S,DK]
            outb[(((size_t)(bb * Hc + hh) * Sc + ss) << 6) + dk] = (bf16)v;
          else             // [B,H,DK,S]
            outb[((size_t)(bb * Hc + hh) * DKc + dk) * Sc + ss] = (bf16)v;
        }
      }
    }
  }
}

// ---------------------------------------------------------------------------
// Fused attention (flash-style, online softmax).  One wave per 16-query tile.
// q,k : [B,H,S,DK] bf16,  vt : [B,H,DK,S] bf16,  ctx out: [B,S,D] bf16
// ---------------------------------------------------------------------------
__global__ void attn_kernel(const bf16* __restrict__ q, const bf16* __restrict__ k,
                            const bf16* __restrict__ vt, bf16* __restrict__ ctx) {
  __shared__ bf16 pbuf[16 * 32];  // P tile staging (C-layout -> A-layout)
  const int lane = threadIdx.x & 31;
  const int hf   = lane >> 4;
  const int n16  = lane & 15;
  const int q0 = blockIdx.x * 16;
  const int bh = blockIdx.y;
  const int bb = bh >> 4;
  const int hh = bh & 15;

  const bf16* qrow  = q  + (((size_t)bh * Sc + q0 + n16) * DKc);
  const bf16* kbase = k  + (size_t)bh * Sc * DKc;
  const bf16* vbase = vt + (size_t)bh * DKc * Sc;

  v16bf aq0 = frag_a(qrow, hf);        // dk 0..31
  v16bf aq1 = frag_a(qrow + 32, hf);   // dk 32..63

  const v8f z = {0.f, 0.f, 0.f, 0.f, 0.f, 0.f, 0.f, 0.f};
  v8f o0 = z, o1 = z, o2 = z, o3 = z;
  float mrow[8], lrow[8];
#pragma unroll
  for (int j = 0; j < 8; ++j) { mrow[j] = -3.0e38f; lrow[j] = 0.f; }

  for (int kb = 0; kb < Sc; kb += 32) {
    const bf16* kr0 = kbase + (size_t)(kb + n16) * DKc;        // keys kb..kb+15
    const bf16* kr1 = kbase + (size_t)(kb + 16 + n16) * DKc;   // keys kb+16..kb+31
    v8f c0 = z, c1 = z;
    c0 = wmma_bf(aq0, frag_b(kr0, hf), c0);
    c0 = wmma_bf(aq1, frag_b(kr0 + 32, hf), c0);
    c1 = wmma_bf(aq0, frag_b(kr1, hf), c1);
    c1 = wmma_bf(aq1, frag_b(kr1 + 32, hf), c1);

#pragma unroll
    for (int j = 0; j < 8; ++j) {
      float s0 = c0[j] * 0.125f;            // 1/sqrt(64)
      float s1 = c1[j] * 0.125f;
      float rmax = redmax16(fmaxf(s0, s1));
      float mn = fmaxf(mrow[j], rmax);
      float sc = __expf(mrow[j] - mn);
      float p0 = __expf(s0 - mn);
      float p1 = __expf(s1 - mn);
      lrow[j] = lrow[j] * sc + redsum16(p0 + p1);
      mrow[j] = mn;
      o0[j] *= sc; o1[j] *= sc; o2[j] *= sc; o3[j] *= sc;
      int r = j + hf * 8;
      pbuf[r * 32 + n16]      = (bf16)p0;
      pbuf[r * 32 + 16 + n16] = (bf16)p1;
    }
    __syncthreads();
    v16bf ap = frag_a(&pbuf[n16 * 32], hf);  // P tile as 16x32 A-fragment
    __syncthreads();

    o0 = wmma_bf(ap, frag_b(vbase + (size_t)(n16)      * Sc + kb, hf), o0);
    o1 = wmma_bf(ap, frag_b(vbase + (size_t)(16 + n16) * Sc + kb, hf), o1);
    o2 = wmma_bf(ap, frag_b(vbase + (size_t)(32 + n16) * Sc + kb, hf), o2);
    o3 = wmma_bf(ap, frag_b(vbase + (size_t)(48 + n16) * Sc + kb, hf), o3);
  }

#pragma unroll
  for (int j = 0; j < 8; ++j) {
    float inv = 1.0f / lrow[j];
    int row = q0 + j + hf * 8;
    size_t base = ((size_t)bb * Sc + row) * Dc + hh * DKc;
    ctx[base + n16]      = (bf16)(o0[j] * inv);
    ctx[base + 16 + n16] = (bf16)(o1[j] * inv);
    ctx[base + 32 + n16] = (bf16)(o2[j] * inv);
    ctx[base + 48 + n16] = (bf16)(o3[j] * inv);
  }
}

// ---------------------------------------------------------------------------
// Host-side orchestration
// in order: x, src_mask(unused by reference), w_q, w_k, w_v, w_o,
//           w1, b1, w2, b2, gamma1, beta1, gamma2, beta2
// ---------------------------------------------------------------------------
extern "C" void kernel_launch(void* const* d_in, const int* in_sizes, int n_in,
                              void* d_out, int out_size, void* d_ws, size_t ws_size,
                              hipStream_t stream) {
  const float* x   = (const float*)d_in[0];
  const float* wq  = (const float*)d_in[2];
  const float* wk  = (const float*)d_in[3];
  const float* wv  = (const float*)d_in[4];
  const float* wo  = (const float*)d_in[5];
  const float* w1  = (const float*)d_in[6];
  const float* b1  = (const float*)d_in[7];
  const float* w2  = (const float*)d_in[8];
  const float* b2  = (const float*)d_in[9];
  const float* g1  = (const float*)d_in[10];
  const float* be1 = (const float*)d_in[11];
  const float* g2  = (const float*)d_in[12];
  const float* be2 = (const float*)d_in[13];

  char* ws = (char*)d_ws;
  size_t off = 0;
  auto take = [&](size_t bytes) {
    char* p = ws + off;
    off += (bytes + 255) & ~(size_t)255;
    return p;
  };
  bf16* wq_b = (bf16*)take((size_t)Dc * Dc * 2);
  bf16* wk_b = (bf16*)take((size_t)Dc * Dc * 2);
  bf16* wv_b = (bf16*)take((size_t)Dc * Dc * 2);
  bf16* wo_b = (bf16*)take((size_t)Dc * Dc * 2);
  bf16* w1_b = (bf16*)take((size_t)DFFc * Dc * 2);
  bf16* w2_b = (bf16*)take((size_t)Dc * DFFc * 2);
  bf16* ln_b = (bf16*)take((size_t)Mtok * Dc * 2);   // reused for LN1 & LN2
  bf16* qb   = (bf16*)take((size_t)Mtok * Dc * 2);   // [B,H,S,DK]
  bf16* kb   = (bf16*)take((size_t)Mtok * Dc * 2);   // [B,H,S,DK]
  bf16* vtb  = (bf16*)take((size_t)Mtok * Dc * 2);   // [B,H,DK,S]
  bf16* ctxb = (bf16*)take((size_t)Mtok * Dc * 2);   // [B,S,D]
  float* x1  = (float*)take((size_t)Mtok * Dc * 4);  // after attn residual
  bf16* h1   = (bf16*)take((size_t)Mtok * DFFc * 2); // FFN hidden
  (void)ws_size; (void)in_sizes; (void)n_in; (void)out_size;

  auto cvt = [&](const float* src, bf16* dst, int n) {
    cvt_bf16_kernel<<<(n + 255) / 256, 256, 0, stream>>>(src, dst, n);
  };
  cvt(wq, wq_b, Dc * Dc);
  cvt(wk, wk_b, Dc * Dc);
  cvt(wv, wv_b, Dc * Dc);
  cvt(wo, wo_b, Dc * Dc);
  cvt(w1, w1_b, DFFc * Dc);
  cvt(w2, w2_b, Dc * DFFc);

  // --- attention block ---
  layernorm_kernel<<<Mtok, 256, 0, stream>>>(x, g1, be1, ln_b);

  dim3 gD(Dc / 64, Mtok / 256);
  gemm_bf16_kernel<<<gD, 256, 0, stream>>>(ln_b, wq_b, nullptr, nullptr, qb, nullptr,
                                           Mtok, Dc, Dc, 0, 2);
  gemm_bf16_kernel<<<gD, 256, 0, stream>>>(ln_b, wk_b, nullptr, nullptr, kb, nullptr,
                                           Mtok, Dc, Dc, 0, 2);
  gemm_bf16_kernel<<<gD, 256, 0, stream>>>(ln_b, wv_b, nullptr, nullptr, vtb, nullptr,
                                           Mtok, Dc, Dc, 0, 3);

  attn_kernel<<<dim3(Sc / 16, Bc * Hc), 32, 0, stream>>>(qb, kb, vtb, ctxb);

  gemm_bf16_kernel<<<gD, 256, 0, stream>>>(ctxb, wo_b, nullptr, x, nullptr, x1,
                                           Mtok, Dc, Dc, 0, 1);

  // --- FFN block ---
  layernorm_kernel<<<Mtok, 256, 0, stream>>>(x1, g2, be2, ln_b);

  dim3 gF(DFFc / 64, Mtok / 256);
  gemm_bf16_kernel<<<gF, 256, 0, stream>>>(ln_b, w1_b, b1, nullptr, h1, nullptr,
                                           Mtok, DFFc, Dc, 1, 0);
  gemm_bf16_kernel<<<gD, 256, 0, stream>>>(h1, w2_b, b2, x1, nullptr, (float*)d_out,
                                           Mtok, Dc, DFFc, 0, 1);
}
